// MultiHeadAttention_70119636074662
// MI455X (gfx1250) — matrix operand order repaired and checked by reading, not compiled, not run on hardware
//
#include <hip/hip_runtime.h>
#include <hip/hip_bf16.h>

// ---------------------------------------------------------------------------
// MHA forward for MI455X (gfx1250), wave32, bf16 WMMA path.
// B=2, L=2048, DM=1024, H=16, DK=DV=64.
// d_out = [context (B,L,DM) fp32][attention (B,H,L,L) fp32]
// ---------------------------------------------------------------------------

typedef __bf16 bf16;
typedef __attribute__((ext_vector_type(16))) __bf16 v16bf;
typedef __attribute__((ext_vector_type(8)))  float  v8f;

#define NB   2
#define NL   2048
#define NDM  1024
#define NH   16
#define NDK  64

__device__ __forceinline__ v8f wmma_bf16(v16bf a, v16bf b, v8f c) {
  // v_wmma_f32_16x16x32_bf16: D = A(16x32) * B(32x16) + C(16x16 f32)
  return __builtin_amdgcn_wmma_f32_16x16x32_bf16(
      /*neg_a=*/false, a, /*neg_b=*/false, b,
      /*c_mod=*/(short)0, c, /*reuse_a=*/false, /*reuse_b=*/false);
}

// A-fragment (16x32, row-major source, convert fp32 -> bf16).
// Lanes 0-15: M=lane, K={0..7,16..23}; lanes 16-31: M=lane-16, K={8..15,24..31}.
__device__ __forceinline__ v16bf a_frag_f32(const float* __restrict__ p0, int ld) {
  int lane = threadIdx.x & 31;
  int m  = lane & 15;
  int kb = (lane & 16) ? 8 : 0;
  const float* p = p0 + (long)m * ld + kb;
  v16bf a;
#pragma unroll
  for (int e = 0; e < 8; ++e) a[e] = (__bf16)p[e];
#pragma unroll
  for (int e = 0; e < 8; ++e) a[e + 8] = (__bf16)p[e + 16];
  return a;
}

__device__ __forceinline__ v16bf a_frag_bf16(const bf16* __restrict__ p0, int ld) {
  int lane = threadIdx.x & 31;
  int m  = lane & 15;
  int kb = (lane & 16) ? 8 : 0;
  const bf16* p = p0 + (long)m * ld + kb;
  v16bf a;
#pragma unroll
  for (int e = 0; e < 8; ++e) a[e] = p[e];
#pragma unroll
  for (int e = 0; e < 8; ++e) a[e + 8] = p[e + 16];
  return a;
}

// B-fragment (32x16): B[k][n] = W[n][k], W row-major along k.
// Lanes 0-15: N=lane, K={0..7,16..23}; lanes 16-31: N=lane-16, K={8..15,24..31}.
__device__ __forceinline__ v16bf b_frag_f32(const float* __restrict__ p0, int ld) {
  int lane = threadIdx.x & 31;
  int n  = lane & 15;
  int kb = (lane & 16) ? 8 : 0;
  const float* p = p0 + (long)n * ld + kb;
  v16bf b;
#pragma unroll
  for (int e = 0; e < 16; ++e) b[e] = (__bf16)p[e < 8 ? e : e + 8];
  return b;
}

__device__ __forceinline__ v16bf b_frag_bf16(const bf16* __restrict__ p0, int ld) {
  int lane = threadIdx.x & 31;
  int n  = lane & 15;
  int kb = (lane & 16) ? 8 : 0;
  const bf16* p = p0 + (long)n * ld + kb;
  v16bf b;
#pragma unroll
  for (int e = 0; e < 16; ++e) b[e] = p[e < 8 ? e : e + 8];
  return b;
}

// ---------------------------------------------------------------------------
// Projection: out = X (M=4096 x K=1024) @ W^T (N=1024 x K=1024), bf16 output.
// transposed==0: out laid (B,H,L,64); transposed==1: out laid (B,H,64,L).
// Each wave: one 16x16 M-tile x 4 N-tiles (A reuse). 512 blocks x 8 waves.
// ---------------------------------------------------------------------------
__global__ __launch_bounds__(256)
void proj_kernel(const float* __restrict__ X, const float* __restrict__ W,
                 bf16* __restrict__ out, int transposed) {
  const int lane = threadIdx.x & 31;
  const int w    = threadIdx.x >> 5;
  const int job  = blockIdx.x * 8 + w;     // 4096 jobs
  const int m0   = (job >> 4) * 16;        // 0..4080
  const int n0   = (job & 15) * 64;        // 0..960

  v8f acc[4] = {};
  for (int k0 = 0; k0 < NDM; k0 += 32) {
    v16bf a = a_frag_f32(X + (long)m0 * NDM + k0, NDM);
#pragma unroll
    for (int t = 0; t < 4; ++t) {
      v16bf b = b_frag_f32(W + (long)(n0 + t * 16) * NDM + k0, NDM);
      acc[t] = wmma_bf16(a, b, acc[t]);
    }
  }

  const int nlane = lane & 15;
  const int mb    = (lane & 16) ? 8 : 0;
  const int bidx  = m0 >> 11;              // batch (tile never crosses batch)
  const int l0    = m0 & (NL - 1);
#pragma unroll
  for (int t = 0; t < 4; ++t) {
    int n = n0 + t * 16 + nlane;
    int h = n >> 6, d = n & 63;
#pragma unroll
    for (int r = 0; r < 8; ++r) {
      int l = l0 + mb + r;
      long idx = transposed
          ? ((long)(bidx * NH + h) * NDK + d) * NL + l
          : ((long)(bidx * NH + h) * NL + l) * NDK + d;
      out[idx] = (__bf16)acc[t][r];
    }
  }
}

// ---------------------------------------------------------------------------
// Attention: per block = (b, h, 16-row query strip).
//   scores (WMMA) -> mask+scale -> LDS -> exact softmax -> stream probs out
//   -> context = P @ V (WMMA, V pre-transposed) -> bf16 ctx (B,L,DM layout).
// ---------------------------------------------------------------------------
#define PLD 2052   // padded LDS row (floats) to break bank conflicts

__global__ __launch_bounds__(256)
void attn_kernel(const bf16* __restrict__ qh, const bf16* __restrict__ kh,
                 const bf16* __restrict__ vhT, const int* __restrict__ mask,
                 float* __restrict__ attn_out, bf16* __restrict__ ctx) {
  __shared__ float s_p[16][PLD];          // 131,328 B score/prob strip
  __shared__ float s_part[8][16][17];     //   8,704 B context partials

  const int lane = threadIdx.x & 31;
  const int w    = threadIdx.x >> 5;
  const int qstrip = blockIdx.x & 127;
  const int h      = (blockIdx.x >> 7) & 15;
  const int b      = blockIdx.x >> 11;
  const int q0     = qstrip * 16;

  const bf16* qhead  = qh  + (long)(b * NH + h) * NL * NDK;
  const bf16* khead  = kh  + (long)(b * NH + h) * NL * NDK;
  const bf16* vthead = vhT + (long)(b * NH + h) * NDK * NL;
  const int*  mrow   = mask + (long)b * NL * NL + (long)q0 * NL;

  const int nlane = lane & 15;
  const int mb    = (lane & 16) ? 8 : 0;

  // ---- Phase 1: scores = scale * Q Kᵀ, masked, into LDS ----
  v16bf aq0 = a_frag_bf16(qhead + (long)q0 * NDK +  0, NDK);
  v16bf aq1 = a_frag_bf16(qhead + (long)q0 * NDK + 32, NDK);
  const float scale = 0.125f;             // 1/sqrt(64)
  for (int kt = w * 16; kt < w * 16 + 16; ++kt) {
    int key0 = kt * 16;
    v16bf bk0 = b_frag_bf16(khead + (long)key0 * NDK +  0, NDK);
    v16bf bk1 = b_frag_bf16(khead + (long)key0 * NDK + 32, NDK);
    v8f c = {};
    c = wmma_bf16(aq0, bk0, c);
    c = wmma_bf16(aq1, bk1, c);
    int n = key0 + nlane;
#pragma unroll
    for (int r = 0; r < 8; ++r) {
      int row  = mb + r;
      int mval = mrow[(long)row * NL + n];
      s_p[row][n] = (mval == 0) ? -1e9f : c[r] * scale;
    }
  }
  __syncthreads();

  // ---- Phase 2: exact softmax, 2 rows per wave ----
#pragma unroll
  for (int rr = 0; rr < 2; ++rr) {
    int row = w * 2 + rr;
    float mx = -3.4e38f;
    for (int cidx = lane; cidx < NL; cidx += 32) mx = fmaxf(mx, s_p[row][cidx]);
#pragma unroll
    for (int off = 16; off; off >>= 1) mx = fmaxf(mx, __shfl_xor(mx, off, 32));
    float sum = 0.f;
    for (int cidx = lane; cidx < NL; cidx += 32) {
      float e = __expf(s_p[row][cidx] - mx);
      s_p[row][cidx] = e;
      sum += e;
    }
#pragma unroll
    for (int off = 16; off; off >>= 1) sum += __shfl_xor(sum, off, 32);
    float inv = 1.0f / sum;
    for (int cidx = lane; cidx < NL; cidx += 32) s_p[row][cidx] *= inv;
  }
  __syncthreads();

  // ---- Phase 3: stream attention probabilities to global (coalesced) ----
  float* arow = attn_out + ((long)(b * NH + h) * NL + q0) * NL;
  for (int idx = threadIdx.x; idx < 16 * NL; idx += 256) {
    int row = idx >> 11, col = idx & (NL - 1);
    arow[(long)row * NL + col] = s_p[row][col];
  }

  // ---- Phase 4: context = P @ V. wave -> (N-tile = w&3, K-half = w>>2) ----
  {
    int n0    = (w & 3) * 16;
    int kbase = (w >> 2) * 1024;
    int m  = lane & 15;
    int kb = (lane & 16) ? 8 : 0;
    v8f c = {};
    for (int k0 = kbase; k0 < kbase + 1024; k0 += 32) {
      const float* p = &s_p[m][k0 + kb];
      v16bf a;
#pragma unroll
      for (int e = 0; e < 8; ++e) a[e] = (__bf16)p[e];
#pragma unroll
      for (int e = 0; e < 8; ++e) a[e + 8] = (__bf16)p[e + 16];
      v16bf bv = b_frag_bf16(vthead + (long)n0 * NL + k0, NL);
      c = wmma_bf16(a, bv, c);
    }
#pragma unroll
    for (int r = 0; r < 8; ++r) s_part[w][mb + r][nlane] = c[r];
  }
  __syncthreads();

  // ---- Phase 5: combine K-halves, store bf16 ctx in (B,L,H*DV) layout ----
  bf16* crow = ctx + ((long)b * NL + q0) * NDM + h * NDK;
  for (int idx = threadIdx.x; idx < 16 * NDK; idx += 256) {
    int r = idx >> 6, d = idx & 63;
    int nt = d >> 4, nn = d & 15;
    float v = s_part[nt][r][nn] + s_part[nt + 4][r][nn];
    crow[(long)r * NDM + d] = (__bf16)v;
  }
}

// ---------------------------------------------------------------------------
// Output projection: out = ctx(bf16, 4096x1024) @ wo_w^T + wo_b  (fp32 out)
// ---------------------------------------------------------------------------
__global__ __launch_bounds__(256)
void outproj_kernel(const bf16* __restrict__ ctxb, const float* __restrict__ Wo,
                    const float* __restrict__ bias, float* __restrict__ out) {
  const int lane = threadIdx.x & 31;
  const int w    = threadIdx.x >> 5;
  const int job  = blockIdx.x * 8 + w;
  const int m0   = (job >> 4) * 16;
  const int n0   = (job & 15) * 64;

  v8f acc[4] = {};
  for (int k0 = 0; k0 < NDM; k0 += 32) {
    v16bf a = a_frag_bf16(ctxb + (long)m0 * NDM + k0, NDM);
#pragma unroll
    for (int t = 0; t < 4; ++t) {
      v16bf b = b_frag_f32(Wo + (long)(n0 + t * 16) * NDM + k0, NDM);
      acc[t] = wmma_bf16(a, b, acc[t]);
    }
  }

  const int nlane = lane & 15;
  const int mb    = (lane & 16) ? 8 : 0;
#pragma unroll
  for (int t = 0; t < 4; ++t) {
    int n = n0 + t * 16 + nlane;
    float bval = bias[n];
#pragma unroll
    for (int r = 0; r < 8; ++r)
      out[(long)(m0 + mb + r) * NDM + n] = acc[t][r] + bval;
  }
}

// ---------------------------------------------------------------------------
extern "C" void kernel_launch(void* const* d_in, const int* in_sizes, int n_in,
                              void* d_out, int out_size, void* d_ws, size_t ws_size,
                              hipStream_t stream) {
  const float* q    = (const float*)d_in[0];
  const float* k    = (const float*)d_in[1];
  const float* v    = (const float*)d_in[2];
  const int*   mask = (const int*)  d_in[3];
  const float* wq   = (const float*)d_in[4];
  const float* wk   = (const float*)d_in[5];
  const float* wv   = (const float*)d_in[6];
  const float* wo_w = (const float*)d_in[7];
  const float* wo_b = (const float*)d_in[8];

  // Workspace layout (bf16): qh | kh | vhT | ctx  -> 4 x 8 MB = 32 MB
  const size_t HEADBYTES = (size_t)NB * NH * NL * NDK * sizeof(bf16);
  char* ws = (char*)d_ws;
  bf16* qh  = (bf16*)(ws);
  bf16* kh  = (bf16*)(ws + HEADBYTES);
  bf16* vhT = (bf16*)(ws + 2 * HEADBYTES);
  bf16* ctx = (bf16*)(ws + 3 * HEADBYTES);

  float* out_ctx  = (float*)d_out;
  float* out_attn = out_ctx + (size_t)NB * NL * NDM;

  dim3 blk(256);
  proj_kernel<<<512, blk, 0, stream>>>(q, wq, qh, 0);
  proj_kernel<<<512, blk, 0, stream>>>(k, wk, kh, 0);
  proj_kernel<<<512, blk, 0, stream>>>(v, wv, vhT, 1);
  attn_kernel<<<NB * NH * (NL / 16), blk, 0, stream>>>(qh, kh, vhT, mask, out_attn, ctx);
  outproj_kernel<<<512, blk, 0, stream>>>(ctx, wo_w, wo_b, out_ctx);
}
